// GNN_Network_3324304687115
// MI455X (gfx1250) — compile-verified, matrix operand order checked by reading, not run on hardware
//
#include <hip/hip_runtime.h>
#include <hip/hip_bf16.h>
#include <math.h>

// ---------------------------------------------------------------------------
// GNN NNConv pipeline for MI455X (gfx1250, wave32).
// Edge MLP GEMM [E,8]x[8,1024] runs on v_wmma_f32_16x16x32_bf16 (K padded
// 8->32 with zeros; bf16 A/B layouts place K=0..7 in lanes 0-15 / elems 0-7).
// tanh + contraction with x[src] fused in registers (never materialize the
// 819MB per-edge weight tensor). Scatter via global f32 atomics.
// ---------------------------------------------------------------------------

typedef __bf16 bf16_t;
typedef __attribute__((ext_vector_type(16))) bf16_t v16bf;
typedef __attribute__((ext_vector_type(8)))  float  v8f;

#define GNN_D  32
#define GNN_FD 8
#define WPB    4      // waves per block in the edge kernel (128 threads)

#if defined(__AMDGCN__) && __has_builtin(__builtin_amdgcn_tanhf)
__device__ __forceinline__ float fast_tanh(float x) { return __builtin_amdgcn_tanhf(x); }
#else
__device__ __forceinline__ float fast_tanh(float x) { return tanhf(x); }
#endif

// ---------------------------------------------------------------------------
// degree: deg[n] = #edges with dst==n
// ---------------------------------------------------------------------------
__global__ void deg_kernel(const int* __restrict__ dstv, float* __restrict__ deg, int E_) {
    int e = blockIdx.x * blockDim.x + threadIdx.x;
    if (e < E_) atomicAdd(&deg[dstv[e]], 1.0f);
}

// ---------------------------------------------------------------------------
// Fused NNConv edge kernel (D->D layers), WMMA bf16 path.
// One wave = one tile of 16 edges. Per tile:
//   A (16x32 bf16) = edge_attr rows, K=0..7 real, rest zero.
//   loop i=0..31 (input channel), oh=0..1 (output half):
//     B (32x16 bf16) from LDS-staged mlpW column chunk,
//     T = wmma(A,B,0); T += bias; T = tanh(T);
//     msg[oh] += xsrc[.,i] * T      (xsrc broadcast from transposed LDS tile)
//   scatter msg tile to agg[dst] with atomics.
// ---------------------------------------------------------------------------
__global__ __launch_bounds__(128) void edge_nnconv_wmma(
    const float* __restrict__ x,
    const int*   __restrict__ src,
    const int*   __restrict__ dstv,
    const float* __restrict__ ea,
    const float* __restrict__ mlpW,   // [8, 1024] row-major
    const float* __restrict__ mlpb,   // [1024]
    float* __restrict__ agg,          // [N, 32] (pre-zeroed)
    int E_, int tilesTotal, int tbMax)
{
    __shared__ bf16_t Wlds[1024 * 8];        // [col][k]  16 KB
    __shared__ float  Blds[1024];            // bias       4 KB
    __shared__ float  Xlds[WPB][GNN_D][20];  // [wave][i][edge] pad->16B-aligned rows

    const int tid = threadIdx.x;
    for (int idx = tid; idx < 1024 * 8; idx += blockDim.x) {
        const int k = idx & 7, col = idx >> 3;
        Wlds[idx] = (bf16_t)mlpW[k * 1024 + col];
    }
    for (int i = tid; i < 1024; i += blockDim.x) Blds[i] = mlpb[i];
    __syncthreads();

    const int wave = tid >> 5;
    const int lane = tid & 31;
    const int half = lane >> 4;   // 0: lanes 0-15, 1: lanes 16-31
    const int r    = lane & 15;

    for (int tb = blockIdx.x; tb < tbMax; tb += gridDim.x) {
        const int  tile = tb * WPB + wave;
        const bool act  = (tile < tilesTotal);
        const int  base = tile * 16;

        // --- A operand: 16 edges x K(=8, zero padded to 32) in bf16 ---
        v16bf a;
        #pragma unroll
        for (int q = 0; q < 16; ++q) a[q] = (bf16_t)0.0f;

        int srcE = 0, dstE = -1;
        if (act && half == 0 && (base + r) < E_) {
            srcE = src[base + r];
            dstE = dstv[base + r];
            const float* ep = ea + (size_t)(base + r) * GNN_FD;
            #pragma unroll
            for (int k = 0; k < GNN_FD; ++k) a[k] = (bf16_t)ep[k];
        }

        // --- stage x[src] tile transposed into LDS: Xlds[i][edge] ---
        const int srcR = __shfl(srcE, r, 32);
        if (act) {
            const float* xr = x + (size_t)srcR * GNN_D + half * 16;
            #pragma unroll
            for (int s = 0; s < 16; ++s)
                Xlds[wave][half * 16 + s][r] = xr[s];
        }
        __syncthreads();

        v8f msg0, msg1;
        #pragma unroll
        for (int q = 0; q < 8; ++q) { msg0[q] = 0.0f; msg1[q] = 0.0f; }

        if (act) {
            for (int i = 0; i < GNN_D; ++i) {
                // xv[j] = x[src[edge j+8*half], i]  (broadcast ds_read, 16B aligned)
                const float* xp = &Xlds[wave][i][8 * half];
                v8f xv;
                #pragma unroll
                for (int q = 0; q < 8; ++q) xv[q] = xp[q];

                #pragma unroll
                for (int oh = 0; oh < 2; ++oh) {
                    const int c = i * 2 + oh;        // 16-col chunk of the 1024-wide MLP
                    v16bf b;
                    #pragma unroll
                    for (int q = 0; q < 16; ++q) b[q] = (bf16_t)0.0f;
                    if (half == 0) {
                        const bf16_t* wp = &Wlds[(c * 16 + r) * 8];
                        #pragma unroll
                        for (int q = 0; q < 8; ++q) b[q] = wp[q];
                    }
                    v8f cz;
                    #pragma unroll
                    for (int q = 0; q < 8; ++q) cz[q] = 0.0f;

                    v8f t = __builtin_amdgcn_wmma_f32_16x16x32_bf16(
                        false, a, false, b, (short)0, cz, false, false);

                    const float bv = Blds[c * 16 + r];
                    if (oh == 0) {
                        #pragma unroll
                        for (int j = 0; j < 8; ++j)
                            msg0[j] += xv[j] * fast_tanh(t[j] + bv);
                    } else {
                        #pragma unroll
                        for (int j = 0; j < 8; ++j)
                            msg1[j] += xv[j] * fast_tanh(t[j] + bv);
                    }
                }
            }
        }
        __syncthreads();   // protect Xlds against next iteration's staging

        // --- scatter msg tile (C layout: row M=j+8*half, col oh*16+r) ---
        if (act) {
            #pragma unroll
            for (int j = 0; j < 8; ++j) {
                const int m = j + 8 * half;
                const int d = __shfl(dstE, m, 32);
                if ((base + m) < E_ && d >= 0) {
                    atomicAdd(&agg[(size_t)d * GNN_D + r],      msg0[j]);
                    atomicAdd(&agg[(size_t)d * GNN_D + 16 + r], msg1[j]);
                }
            }
        }
    }
}

// ---------------------------------------------------------------------------
// h = agg/deg + x@root + bias; also accumulate per-column sum / sumsq for BN.
// ---------------------------------------------------------------------------
__global__ __launch_bounds__(256) void node_h_kernel(
    const float* __restrict__ agg, const float* __restrict__ deg,
    const float* __restrict__ x,   const float* __restrict__ root,
    const float* __restrict__ bias, float* __restrict__ h,
    float* __restrict__ stats, int Nn)
{
    __shared__ float rootS[GNN_D * GNN_D];
    __shared__ float ssum[GNN_D], ssq[GNN_D];
    const int tid = threadIdx.x;
    for (int i = tid; i < GNN_D * GNN_D; i += 256) rootS[i] = root[i];
    if (tid < GNN_D) { ssum[tid] = 0.0f; ssq[tid] = 0.0f; }
    __syncthreads();

    const int g = blockIdx.x * 256 + tid;
    const int n = g >> 5, o = g & 31;
    float hv = 0.0f;
    if (n < Nn) {
        float d = deg[n]; d = d > 1.0f ? d : 1.0f;
        float acc = agg[g] / d + bias[o];
        const float* xr = x + (size_t)n * GNN_D;
        #pragma unroll
        for (int i = 0; i < GNN_D; ++i) acc += xr[i] * rootS[i * GNN_D + o];
        hv = acc;
        h[g] = acc;
    }
    atomicAdd(&ssum[o], hv);
    atomicAdd(&ssq[o], hv * hv);
    __syncthreads();
    if (tid < GNN_D) {
        atomicAdd(&stats[tid],          ssum[tid]);
        atomicAdd(&stats[GNN_D + tid],  ssq[tid]);
    }
}

// ---------------------------------------------------------------------------
// x_out = x + relu( gamma*(h-mu)*rsqrt(var+eps) + beta ), biased var.
// ---------------------------------------------------------------------------
__global__ void bn_apply_kernel(
    const float* __restrict__ x, const float* __restrict__ h,
    const float* __restrict__ stats,
    const float* __restrict__ gamma, const float* __restrict__ beta,
    float* __restrict__ xout, int Nn)
{
    const int g = blockIdx.x * blockDim.x + threadIdx.x;
    if (g >= Nn * GNN_D) return;
    const int o = g & 31;
    const float invN = 1.0f / (float)Nn;
    const float mu  = stats[o] * invN;
    const float var = stats[GNN_D + o] * invN - mu * mu;
    const float hn  = gamma[o] * (h[g] - mu) * rsqrtf(var + 1e-5f) + beta[o];
    xout[g] = x[g] + (hn > 0.0f ? hn : 0.0f);
}

// ---------------------------------------------------------------------------
// Final NNConv (D -> 1): per-edge 32-wide MLP + dot with x[src], scalar scatter.
// ---------------------------------------------------------------------------
__global__ void edge_final_kernel(
    const float* __restrict__ x, const int* __restrict__ src,
    const int* __restrict__ dstv, const float* __restrict__ ea,
    const float* __restrict__ mlpWf, const float* __restrict__ mlpbf,
    float* __restrict__ aggf, int E_)
{
    const int e = blockIdx.x * blockDim.x + threadIdx.x;
    if (e >= E_) return;
    float eav[GNN_FD];
    #pragma unroll
    for (int k = 0; k < GNN_FD; ++k) eav[k] = ea[(size_t)e * GNN_FD + k];
    const float* xr = x + (size_t)src[e] * GNN_D;
    float msg = 0.0f;
    #pragma unroll 4
    for (int o = 0; o < GNN_D; ++o) {
        float acc = mlpbf[o];
        #pragma unroll
        for (int k = 0; k < GNN_FD; ++k) acc += eav[k] * mlpWf[k * GNN_D + o];
        msg += xr[o] * fast_tanh(acc);
    }
    atomicAdd(&aggf[dstv[e]], msg);
}

__global__ void out_kernel(
    const float* __restrict__ aggf, const float* __restrict__ deg,
    const float* __restrict__ x, const float* __restrict__ rootf,
    const float* __restrict__ biasf, float* __restrict__ out, int Nn)
{
    const int n = blockIdx.x * blockDim.x + threadIdx.x;
    if (n >= Nn) return;
    float d = deg[n]; d = d > 1.0f ? d : 1.0f;
    float acc = aggf[n] / d + biasf[0];
    const float* xr = x + (size_t)n * GNN_D;
    #pragma unroll
    for (int i = 0; i < GNN_D; ++i) acc += xr[i] * rootf[i];
    out[n] = acc;
}

// ---------------------------------------------------------------------------
extern "C" void kernel_launch(void* const* d_in, const int* in_sizes, int n_in,
                              void* d_out, int out_size, void* d_ws, size_t ws_size,
                              hipStream_t stream) {
    (void)n_in; (void)out_size; (void)ws_size;
    const float* x0    = (const float*)d_in[0];
    const int*   ei    = (const int*)  d_in[1];
    const float* ea    = (const float*)d_in[2];
    const float* mlpW0 = (const float*)d_in[3];
    const float* mlpb0 = (const float*)d_in[4];
    const float* root0 = (const float*)d_in[5];
    const float* bias0 = (const float*)d_in[6];
    const float* gam0  = (const float*)d_in[7];
    const float* bet0  = (const float*)d_in[8];
    const float* mlpW1 = (const float*)d_in[9];
    const float* mlpb1 = (const float*)d_in[10];
    const float* root1 = (const float*)d_in[11];
    const float* bias1 = (const float*)d_in[12];
    const float* gam1  = (const float*)d_in[13];
    const float* bet1  = (const float*)d_in[14];
    const float* mlpWf = (const float*)d_in[15];
    const float* mlpbf = (const float*)d_in[16];
    const float* rootf = (const float*)d_in[17];
    const float* biasf = (const float*)d_in[18];

    const int Nn = in_sizes[0] / GNN_D;
    const int Ee = in_sizes[1] / 2;
    const int* src  = ei;
    const int* dstv = ei + Ee;

    float* ws    = (float*)d_ws;
    float* deg   = ws;  ws += Nn;
    float* agg   = ws;  ws += (size_t)Nn * GNN_D;
    float* h     = ws;  ws += (size_t)Nn * GNN_D;
    float* x1    = ws;  ws += (size_t)Nn * GNN_D;
    float* x2    = ws;  ws += (size_t)Nn * GNN_D;
    float* stats = ws;  ws += 64;
    float* aggf  = ws;  ws += Nn;

    const int tiles = (Ee + 15) / 16;
    const int tbMax = (tiles + WPB - 1) / WPB;
    dim3 egrid(tbMax), eblk(WPB * 32);
    const int nth = Nn * GNN_D;
    dim3 ngrid((nth + 255) / 256), nblk(256);

    hipMemsetAsync(deg, 0, sizeof(float) * (size_t)Nn, stream);
    deg_kernel<<<(Ee + 255) / 256, 256, 0, stream>>>(dstv, deg, Ee);

    // ---- layer 0 ----
    hipMemsetAsync(agg,   0, sizeof(float) * (size_t)Nn * GNN_D, stream);
    hipMemsetAsync(stats, 0, sizeof(float) * 64, stream);
    edge_nnconv_wmma<<<egrid, eblk, 0, stream>>>(x0, src, dstv, ea, mlpW0, mlpb0,
                                                 agg, Ee, tiles, tbMax);
    node_h_kernel<<<ngrid, nblk, 0, stream>>>(agg, deg, x0, root0, bias0, h, stats, Nn);
    bn_apply_kernel<<<ngrid, nblk, 0, stream>>>(x0, h, stats, gam0, bet0, x1, Nn);

    // ---- layer 1 ----
    hipMemsetAsync(agg,   0, sizeof(float) * (size_t)Nn * GNN_D, stream);
    hipMemsetAsync(stats, 0, sizeof(float) * 64, stream);
    edge_nnconv_wmma<<<egrid, eblk, 0, stream>>>(x1, src, dstv, ea, mlpW1, mlpb1,
                                                 agg, Ee, tiles, tbMax);
    node_h_kernel<<<ngrid, nblk, 0, stream>>>(agg, deg, x1, root1, bias1, h, stats, Nn);
    bn_apply_kernel<<<ngrid, nblk, 0, stream>>>(x1, h, stats, gam1, bet1, x2, Nn);

    // ---- final D -> 1 ----
    hipMemsetAsync(aggf, 0, sizeof(float) * (size_t)Nn, stream);
    edge_final_kernel<<<(Ee + 255) / 256, 256, 0, stream>>>(x2, src, dstv, ea,
                                                            mlpWf, mlpbf, aggf, Ee);
    out_kernel<<<(Nn + 255) / 256, 256, 0, stream>>>(aggf, deg, x2, rootf, biasf,
                                                     (float*)d_out, Nn);
}
